// RefineFullyDeformableCrossAttentionTransformer_1898375545238
// MI455X (gfx1250) — matrix operand order repaired and checked by reading, not compiled, not run on hardware
//
#include <hip/hip_runtime.h>
#include <hip/hip_bf16.h>
#include <math.h>

// ---------------- problem constants ----------------
#define NB      4
#define LQ      900
#define DIM     256
#define HEADS   8
#define DH      32
#define NPTS    4
#define HW      128
#define LEN     (HW*HW)       // 16384
#define DFF     1024
#define NLAYERS 6
#define M1      (NB*LQ)       // 3600 (multiple of 16)
#define MV      (NB*LEN)      // 65536

typedef __attribute__((ext_vector_type(16))) __bf16 v16bf;
typedef __attribute__((ext_vector_type(8)))  float  v8f;

static __device__ __forceinline__ __bf16 to_bf16(float f) { return (__bf16)f; }

// ---- CDNA5 async global->LDS path (guarded; sync fallback keeps file compiling) ----
#if defined(__has_builtin)
#if __has_builtin(__builtin_amdgcn_global_load_async_to_lds_b128)
#define HAVE_ASYNC_LDS 1
#endif
#if __has_builtin(__builtin_amdgcn_s_wait_asynccnt)
#define HAVE_WAIT_ASYNC 1
#endif
#endif

#ifdef HAVE_ASYNC_LDS
// builtin signature (from clang diagnostic): arg0 = addrspace(1) ptr to
// 'int __attribute__((vector_size(16)))'; assume matching addrspace(3) for arg1.
typedef int v4i_vs __attribute__((vector_size(16)));
typedef __attribute__((address_space(1))) v4i_vs* glb_v4i_p;
typedef __attribute__((address_space(3))) v4i_vs* lds_v4i_p;
static __device__ __forceinline__ void async_cp16(const float* g, float* l) {
    __builtin_amdgcn_global_load_async_to_lds_b128((glb_v4i_p)(g), (lds_v4i_p)(l), 0, 0);
}
static __device__ __forceinline__ void wait_async0() {
#ifdef HAVE_WAIT_ASYNC
    __builtin_amdgcn_s_wait_asynccnt(0);
#else
    asm volatile("s_wait_asynccnt 0" ::: "memory");
#endif
}
#else
static __device__ __forceinline__ void async_cp16(const float* g, float* l) {
    *(float4*)l = *(const float4*)g;
}
static __device__ __forceinline__ void wait_async0() {}
#endif

// ---------------------------------------------------------------------------
// Tiled GEMM: C[M,N] = A[M,K] @ B[K,N] + bias (opt ReLU / row-mask).
// 128-thread block (4 waves) -> 64x64 C tile; K in 32-chunks, double-buffered
// async staging into LDS; each wave: one 16-row strip x 4 N-tiles (4 WMMA/chunk).
// Requires: K%32==0, N%64==0, M%16==0.
// ---------------------------------------------------------------------------
#define BK 32
template <bool RELU, bool MASK>
__global__ __launch_bounds__(128)
void gemm_tiled_kernel(const float* __restrict__ A, const float* __restrict__ B,
                       const float* __restrict__ bias, float* __restrict__ C,
                       int M, int Nn, int K, const unsigned char* __restrict__ mask)
{
    __shared__ float As[2][64 * BK];   // rows m, cols k   (8 KB each)
    __shared__ float Bs[2][BK * 64];   // rows k, cols n   (8 KB each)

    const int t    = threadIdx.x;
    const int wv   = t >> 5;
    const int lane = t & 31;
    const int half = lane >> 4;
    const int r    = lane & 15;
    const int m0b  = blockIdx.x * 64;
    const int n0b  = blockIdx.y * 64;
    const int nchunks = K / BK;

    // staging assignment: A: thread -> (row t/2, 16-col seg t%2); B: (row t/4, seg t%4)
    const int arow = t >> 1, aseg = (t & 1) << 4;
    const int brow = t >> 2, bseg = (t & 3) << 4;
    const int garow = (m0b + arow < M) ? (m0b + arow) : (M - 1);   // clamp (M%64 may be !=0)

    v8f acc[4] = {};

    // prologue: stage chunk 0 into buffer 0
    {
        const float* ga = A + (size_t)garow * K + aseg;
        const float* gb = B + (size_t)brow * Nn + n0b + bseg;
        float* la = &As[0][arow * BK + aseg];
        float* lb = &Bs[0][brow * 64 + bseg];
#pragma unroll
        for (int i = 0; i < 4; ++i) async_cp16(ga + 4 * i, la + 4 * i);
#pragma unroll
        for (int i = 0; i < 4; ++i) async_cp16(gb + 4 * i, lb + 4 * i);
    }

    for (int c = 0; c < nchunks; ++c) {
        const int buf = c & 1;
        wait_async0();
        __syncthreads();                        // staged data visible to all waves

        if (c + 1 < nchunks) {                  // stage next chunk into other buffer
            const int k = (c + 1) * BK;
            const float* ga = A + (size_t)garow * K + k + aseg;
            const float* gb = B + (size_t)(k + brow) * Nn + n0b + bseg;
            float* la = &As[buf ^ 1][arow * BK + aseg];
            float* lb = &Bs[buf ^ 1][brow * 64 + bseg];
#pragma unroll
            for (int i = 0; i < 4; ++i) async_cp16(ga + 4 * i, la + 4 * i);
#pragma unroll
            for (int i = 0; i < 4; ++i) async_cp16(gb + 4 * i, lb + 4 * i);
        }

        // A fragment (reused for 4 WMMAs): elem e -> K=half*16+e
        const float* abase = &As[buf][(wv * 16 + r) * BK];
        v16bf af;
#pragma unroll
        for (int e = 0; e < 8; ++e) {
            af[e]     = to_bf16(abase[half * 16 + e]);
            af[e + 8] = to_bf16(abase[half * 16 + 8 + e]);
        }
#pragma unroll
        for (int j = 0; j < 4; ++j) {           // B fragment: elem e -> K=half*16+e, col j*16+r
            const float* bbase = &Bs[buf][(half * 16) * 64 + j * 16 + r];
            v16bf bfv;
#pragma unroll
            for (int e = 0; e < 16; ++e) bfv[e] = to_bf16(bbase[e * 64]);
            acc[j] = __builtin_amdgcn_wmma_f32_16x16x32_bf16(
                         false, af, false, bfv, (short)0, acc[j], false, false);
        }
    }

    if (m0b + wv * 16 >= M) return;             // wave-uniform (M%16==0)
#pragma unroll
    for (int j = 0; j < 4; ++j) {
        const int   col = n0b + j * 16 + r;
        const float bv  = bias[col];
#pragma unroll
        for (int v = 0; v < 8; ++v) {           // C layout: row = half*8 + v
            const int row = m0b + wv * 16 + (half << 3) + v;
            float val = acc[j][v] + bv;
            if (RELU) val = fmaxf(val, 0.f);
            if (MASK) { if (mask[row]) val = 0.f; }
            C[(size_t)row * Nn + col] = val;
        }
    }
}

// ---------------------------------------------------------------------------
// Per-wave GEMM for narrow N (N=32): one wave -> one 16x16 tile.
// ---------------------------------------------------------------------------
__global__ void gemm_wave_kernel(const float* __restrict__ A, const float* __restrict__ B,
                                 const float* __restrict__ bias, float* __restrict__ C,
                                 int M, int Nn, int K)
{
    const int wave   = (int)((blockIdx.x * blockDim.x + threadIdx.x) >> 5);
    const int ntiles = Nn >> 4;
    const int mt     = wave / ntiles;
    const int nt     = wave - mt * ntiles;
    if (mt >= (M >> 4)) return;

    const int lane = threadIdx.x & 31;
    const int half = lane >> 4, r = lane & 15;
    const int m0 = mt << 4, n0 = nt << 4;

    v8f acc = {};
    for (int k = 0; k < K; k += 32) {
        v16bf af, bfv;
        const float* arow = A + (size_t)(m0 + r) * K + k + half * 16;
#pragma unroll
        for (int e = 0; e < 8; ++e) { af[e] = to_bf16(arow[e]); af[e + 8] = to_bf16(arow[e + 8]); }
        const float* bcol = B + (size_t)(k + half * 16) * Nn + n0 + r;
#pragma unroll
        for (int e = 0; e < 16; ++e) bfv[e] = to_bf16(bcol[(size_t)e * Nn]);
        acc = __builtin_amdgcn_wmma_f32_16x16x32_bf16(false, af, false, bfv, (short)0, acc, false, false);
    }
    const int col = n0 + r;
    const float bv = bias[col];
#pragma unroll
    for (int v = 0; v < 8; ++v)
        C[(size_t)(m0 + (half << 3) + v) * Nn + col] = acc[v] + bv;
}

// ---------------------------------------------------------------------------
__global__ void add2_kernel(const float* __restrict__ a, const float* __restrict__ b,
                            float* __restrict__ o, int n)
{
    int i = blockIdx.x * blockDim.x + threadIdx.x;
    if (i < n) o[i] = a[i] + b[i];
}

__global__ void softmax32_kernel(float* __restrict__ aw, int rows)
{
    int r = blockIdx.x * blockDim.x + threadIdx.x;
    if (r >= rows) return;
    float* p = aw + (size_t)r * 32;
    float m = -1e30f;
#pragma unroll
    for (int i = 0; i < 32; ++i) m = fmaxf(m, p[i]);
    float e[32], s = 0.f;
#pragma unroll
    for (int i = 0; i < 32; ++i) { e[i] = expf(p[i] - m); s += e[i]; }
    float inv = 1.f / s;
#pragma unroll
    for (int i = 0; i < 32; ++i) p[i] = e[i] * inv;
}

// out = LN(a + b) * g + beta   (256-thread block per row)
__global__ void add_ln_kernel(const float* __restrict__ a, const float* __restrict__ b,
                              const float* __restrict__ g, const float* __restrict__ bt,
                              float* __restrict__ out)
{
    const int row = blockIdx.x, t = threadIdx.x;
    const size_t i = (size_t)row * DIM + t;
    float v = a[i] + b[i];
    __shared__ float red[DIM];
    red[t] = v; __syncthreads();
    for (int s = DIM / 2; s > 0; s >>= 1) { if (t < s) red[t] += red[t + s]; __syncthreads(); }
    const float mean = red[0] * (1.f / DIM);
    __syncthreads();
    const float d = v - mean;
    red[t] = d * d; __syncthreads();
    for (int s = DIM / 2; s > 0; s >>= 1) { if (t < s) red[t] += red[t + s]; __syncthreads(); }
    const float var = red[0] * (1.f / DIM);
    out[i] = d * rsqrtf(var + 1e-5f) * g[t] + bt[t];
}

// ---------------------------------------------------------------------------
// Deformable sampling + attention-weighted sum: one wave per (n,q,head), lane=channel.
// ---------------------------------------------------------------------------
__global__ void sample_attn_kernel(const float* __restrict__ value,   // (N,LEN,D)
                                   const float* __restrict__ refp,    // (N,LQ,4)
                                   const float* __restrict__ svr,     // (N,1,2)
                                   const float* __restrict__ off,     // (N*LQ,64)
                                   const float* __restrict__ aw,      // (N*LQ,32)
                                   float* __restrict__ attn)          // (N*LQ,256)
{
    const int wid = (int)((blockIdx.x * blockDim.x + threadIdx.x) >> 5);
    if (wid >= M1 * HEADS) return;
    const int h  = wid % HEADS;
    const int nq = wid / HEADS;
    const int n  = nq / LQ;
    const int d  = threadIdx.x & 31;

    const float vrx = svr[n * 2 + 0], vry = svr[n * 2 + 1];
    const float rx = refp[(size_t)nq * 4 + 0] * vrx;
    const float ry = refp[(size_t)nq * 4 + 1] * vry;
    const float rw = refp[(size_t)nq * 4 + 2] * vrx;
    const float rh = refp[(size_t)nq * 4 + 3] * vry;

    const float* vbase = value + (size_t)n * LEN * DIM + h * DH + d;
    float acc = 0.f;
#pragma unroll
    for (int p = 0; p < NPTS; ++p) {
        const float ox = off[(size_t)nq * 64 + h * 8 + p * 2 + 0];
        const float oy = off[(size_t)nq * 64 + h * 8 + p * 2 + 1];
        const float gx = (rx + ox * 0.125f * rw) * (float)HW - 0.5f;   // off/NPTS * w * 0.5
        const float gy = (ry + oy * 0.125f * rh) * (float)HW - 0.5f;
        const float x0 = floorf(gx), y0 = floorf(gy);
        const float wx = gx - x0,   wy = gy - y0;
        const int xi = (int)x0, yi = (int)y0;

        float s = 0.f;
#pragma unroll
        for (int cy = 0; cy < 2; ++cy) {
#pragma unroll
            for (int cx = 0; cx < 2; ++cx) {
                const int xx = xi + cx, yy = yi + cy;
                const float w = (cx ? wx : 1.f - wx) * (cy ? wy : 1.f - wy);
                if (xx >= 0 && xx < HW && yy >= 0 && yy < HW)
                    s += vbase[(size_t)(yy * HW + xx) * DIM] * w;
            }
        }
        acc += s * aw[(size_t)nq * 32 + h * 4 + p];
    }
    attn[(size_t)nq * DIM + h * DH + d] = acc;
}

// bbox head: C[M,4] = A[M,256] @ B[256,4] + bias
__global__ void bbox_head_kernel(const float* __restrict__ A, const float* __restrict__ B,
                                 const float* __restrict__ bias, float* __restrict__ C, int M)
{
    int i = blockIdx.x * blockDim.x + threadIdx.x;
    if (i >= M * 4) return;
    const int row = i >> 2, c = i & 3;
    float s = bias[c];
    const float* a = A + (size_t)row * DIM;
    for (int k = 0; k < DIM; ++k) s += a[k] * B[(size_t)k * 4 + c];
    C[i] = s;
}

// ref_new = sigmoid(tmp + inv_sigmoid(clip(ref)))
__global__ void refine_kernel(const float* __restrict__ tmp, const float* __restrict__ refin,
                              float* __restrict__ refout, int n)
{
    int i = blockIdx.x * blockDim.x + threadIdx.x;
    if (i >= n) return;
    float r = fminf(fmaxf(refin[i], 0.f), 1.f);
    const float inv = logf(fmaxf(r, 1e-5f) / fmaxf(1.f - r, 1e-5f));
    refout[i] = 1.f / (1.f + expf(-(tmp[i] + inv)));
}

// ---------------------------------------------------------------------------
static inline void launch_gemm(bool relu, bool maskEn,
                               const float* A, const float* B, const float* bias, float* C,
                               int M, int Nn, int K, const unsigned char* mask,
                               hipStream_t stream)
{
    if ((Nn & 63) == 0) {
        dim3 grid((M + 63) / 64, Nn / 64);
        if (relu)
            gemm_tiled_kernel<true , false><<<grid, 128, 0, stream>>>(A, B, bias, C, M, Nn, K, mask);
        else if (maskEn)
            gemm_tiled_kernel<false, true ><<<grid, 128, 0, stream>>>(A, B, bias, C, M, Nn, K, mask);
        else
            gemm_tiled_kernel<false, false><<<grid, 128, 0, stream>>>(A, B, bias, C, M, Nn, K, mask);
    } else {
        const long waves  = (long)(M >> 4) * (Nn >> 4);
        const int  blocks = (int)((waves + 3) / 4);
        gemm_wave_kernel<<<blocks, 128, 0, stream>>>(A, B, bias, C, M, Nn, K);
    }
}

extern "C" void kernel_launch(void* const* d_in, const int* in_sizes, int n_in,
                              void* d_out, int out_size, void* d_ws, size_t ws_size,
                              hipStream_t stream)
{
    (void)in_sizes; (void)n_in; (void)out_size; (void)ws_size;

    const float* tgt   = (const float*)d_in[0];
    const float* refpt = (const float*)d_in[1];
    const float* src   = (const float*)d_in[2];
    const float* svr   = (const float*)d_in[3];
    const float* qpos  = (const float*)d_in[4];
    const float* Wo    = (const float*)d_in[5];
    const float* bo    = (const float*)d_in[6];
    const float* Wa    = (const float*)d_in[7];
    const float* ba    = (const float*)d_in[8];
    const float* Wv    = (const float*)d_in[9];
    const float* bv    = (const float*)d_in[10];
    const float* Wout  = (const float*)d_in[11];
    const float* bout  = (const float*)d_in[12];
    const float* g1    = (const float*)d_in[13];
    const float* b1    = (const float*)d_in[14];
    const float* Wfc   = (const float*)d_in[15];
    const float* bfc   = (const float*)d_in[16];
    const float* Wproj = (const float*)d_in[17];
    const float* bproj = (const float*)d_in[18];
    const float* g2    = (const float*)d_in[19];
    const float* b2    = (const float*)d_in[20];
    const float* Wb1   = (const float*)d_in[21];
    const float* bb1   = (const float*)d_in[22];
    const float* Wb2   = (const float*)d_in[23];
    const float* bb2   = (const float*)d_in[24];
    const float* Wb3   = (const float*)d_in[25];
    const float* bb3   = (const float*)d_in[26];
    const unsigned char* mask = (const unsigned char*)d_in[29];

    float* out = (float*)d_out;
    const size_t SZ      = (size_t)M1 * DIM;
    const size_t REFBASE = (size_t)NLAYERS * SZ;

    float* ws    = (float*)d_ws;
    float* vbuf  = ws;                               // MV*DIM
    float* qbuf  = vbuf  + (size_t)MV * DIM;
    float* offb  = qbuf  + SZ;
    float* awb   = offb  + (size_t)M1 * 64;
    float* attnb = awb   + (size_t)M1 * 32;
    float* projb = attnb + SZ;                       // also reused as t2
    float* xb    = projb + SZ;
    float* hb    = xb    + SZ;
    float* t1b   = hb    + (size_t)M1 * DFF;
    float* tmpb  = t1b   + SZ;

    for (int l = 0; l < NLAYERS; ++l) {
        const float* outp = (l == 0) ? tgt   : out + (size_t)(l - 1) * SZ;
        const float* refp = (l == 0) ? refpt : out + REFBASE + (size_t)(l - 1) * M1 * 4;
        float* xout   = out + (size_t)l * SZ;
        float* refout = out + REFBASE + (size_t)l * M1 * 4;

        // value = where(mask, 0, src @ Wv[l] + bv[l])     (65536 x 256 x 256)
        launch_gemm(false, true, src, Wv + (size_t)l * DIM * DIM, bv + l * DIM,
                    vbuf, MV, DIM, DIM, mask, stream);

        // q = output + query_pos
        add2_kernel<<<(int)((SZ + 255) / 256), 256, 0, stream>>>(outp, qpos, qbuf, (int)SZ);

        // sampling offsets (N=64, tiled) and attention weights (N=32, per-wave) + softmax
        launch_gemm(false, false, qbuf, Wo + (size_t)l * DIM * 64, bo + l * 64,
                    offb, M1, 64, DIM, nullptr, stream);
        launch_gemm(false, false, qbuf, Wa + (size_t)l * DIM * 32, ba + l * 32,
                    awb, M1, 32, DIM, nullptr, stream);
        softmax32_kernel<<<(M1 + 127) / 128, 128, 0, stream>>>(awb, M1);

        // bilinear sampling + weighted sum over points
        sample_attn_kernel<<<(M1 * HEADS * 32 + 127) / 128, 128, 0, stream>>>(
            vbuf, refp, svr, offb, awb, attnb);

        // output projection + residual + LN1
        launch_gemm(false, false, attnb, Wout + (size_t)l * DIM * DIM, bout + l * DIM,
                    projb, M1, DIM, DIM, nullptr, stream);
        add_ln_kernel<<<M1, DIM, 0, stream>>>(outp, projb, g1 + l * DIM, b1 + l * DIM, xb);

        // FFN + residual + LN2 -> inter[l]
        launch_gemm(true,  false, xb, Wfc + (size_t)l * DIM * DFF, bfc + l * DFF,
                    hb, M1, DFF, DIM, nullptr, stream);
        launch_gemm(false, false, hb, Wproj + (size_t)l * DFF * DIM, bproj + l * DIM,
                    projb, M1, DIM, DFF, nullptr, stream);
        add_ln_kernel<<<M1, DIM, 0, stream>>>(xb, projb, g2 + l * DIM, b2 + l * DIM, xout);

        // bbox refinement MLP (256 -> 256 -> 256 -> 4) on inter[l]
        launch_gemm(true, false, xout, Wb1 + (size_t)l * DIM * DIM, bb1 + l * DIM,
                    t1b, M1, DIM, DIM, nullptr, stream);
        launch_gemm(true, false, t1b, Wb2 + (size_t)l * DIM * DIM, bb2 + l * DIM,
                    projb, M1, DIM, DIM, nullptr, stream);
        bbox_head_kernel<<<(M1 * 4 + 127) / 128, 128, 0, stream>>>(
            projb, Wb3 + (size_t)l * DIM * 4, bb3 + l * 4, tmpb, M1);
        refine_kernel<<<(M1 * 4 + 127) / 128, 128, 0, stream>>>(tmpb, refp, refout, M1 * 4);
    }
}
